// cheb_conv_withSAt_26757646254628
// MI455X (gfx1250) — matrix operand order repaired and checked by reading, not compiled, not run on hardware
//
#include <hip/hip_runtime.h>

// ---------------------------------------------------------------------------
// ChebConv with spatial attention, bf16 WMMA path for gfx1250 (MI455X).
// B=16, N=2048, F_IN=32, T=24, K=3, F_OUT=64
//   rhs_k[b,m,f,t] = sum_n (cheb_k[n,m]*att[b,n,m]) * x[b,n,f,t]
//   out[b,m,o,t]   = relu( sum_k sum_f rhs_k[b,m,f,t] * Theta[k,f,o] )
//
// M=32 per workgroup (512 thr / 16 waves); each wave owns both m-tiles so
// every X B-fragment feeds two WMMAs; register-side software pipeline on the
// cheb*att tile build; contiguous 32B fragments loaded as single v16bf.
// ---------------------------------------------------------------------------

typedef __attribute__((ext_vector_type(16))) __bf16 v16bf;
typedef __attribute__((ext_vector_type(8)))  __bf16 v8bf;
typedef __attribute__((ext_vector_type(8)))  float  v8f;

constexpr int Bn   = 16;
constexpr int Nn   = 2048;
constexpr int FIN  = 32;
constexpr int Tt   = 24;
constexpr int Kk   = 3;
constexpr int FOUT = 64;
constexpr int TF   = FIN * Tt;   // 768, also Tt*32
constexpr int Mw   = 32;         // m rows per workgroup

constexpr int AST = 40;          // sAt row stride in halfs (80B, 16B-aligned rows)
constexpr int RST = TF + 8;      // sRhs row stride in halfs (1552B, 16B-aligned)

// ---------------------------------------------------------------------------
// Stage 0: x (B,N,F_IN,T) fp32 -> Xt (B, TF, N) bf16 where tf = t*32 + f.
// Puts the contraction axis n contiguous for WMMA B-fragment loads.
// ---------------------------------------------------------------------------
__global__ __launch_bounds__(256) void xpose_bf16_kernel(
    const float* __restrict__ x, __bf16* __restrict__ Xt) {
  __shared__ float tile[32][33];
  const int tx = threadIdx.x;           // 0..31
  const int ty = threadIdx.y;           // 0..7
  const int n0  = blockIdx.x * 32;
  const int ft0 = blockIdx.y * 32;      // ft = f*24 + t (input-contiguous)
  const int b   = blockIdx.z;

  const float* xin = x + ((size_t)b * Nn) * TF;
#pragma unroll
  for (int r = 0; r < 4; ++r) {
    int nl = ty + r * 8;
    tile[nl][tx] = xin[(size_t)(n0 + nl) * TF + (ft0 + tx)];   // coalesced read
  }
  __syncthreads();
#pragma unroll
  for (int r = 0; r < 4; ++r) {
    int ftl = ty + r * 8;
    int ft  = ft0 + ftl;
    int f   = ft / Tt;
    int t   = ft - f * Tt;
    int tf  = t * FIN + f;                                      // transposed index
    Xt[((size_t)b * TF + tf) * Nn + (n0 + tx)] = (__bf16)tile[tx][ftl]; // coalesced write
  }
}

// ---------------------------------------------------------------------------
// Main fused kernel: one workgroup (16 waves) per (b, 32-row m-tile).
// Wave w: tf slice = w*48 (3 tiles of 16), handles BOTH 16-row m-tiles.
// ---------------------------------------------------------------------------
__global__ __launch_bounds__(512, 1) void cheb_conv_wmma_kernel(
    const float* __restrict__ att,    // (B,N,N)
    const float* __restrict__ cheb,   // (K,N,N)
    const float* __restrict__ Theta,  // (K,F_IN,F_OUT)
    const __bf16* __restrict__ Xt,    // (B,TF,N) bf16
    float* __restrict__ out) {        // (B,N,F_OUT,T)

  __shared__ __align__(16) __bf16 sAt[Mw * AST];            // A^T tile  (2.5 KB)
  __shared__ __align__(32) __bf16 sTheta[Kk * FOUT * FIN];  // [k][o][f] (12 KB)
  __shared__ __align__(16) unsigned char sBig[Mw * RST * 2];// rhs / out staging (48.5 KB)
  __bf16* sRhs  = (__bf16*)sBig;                            // [m][tf], stride RST
  float*  sOutC = (float*)sBig;                             // 32*16*24 fp32 chunk

  const int tid  = threadIdx.x;
  const int wave = tid >> 5;      // 0..15
  const int lane = tid & 31;
  const int h    = lane >> 4;     // lane half (WMMA layout)
  const int l16  = lane & 15;
  const int b    = blockIdx.y;
  const int m0   = blockIdx.x * Mw;
  const int tfbase = wave * 48;   // 3 tiles of 16 tf columns per wave

  // Theta -> LDS as bf16 [k][o][f] (B-fragment friendly: f contiguous)
  for (int i = tid; i < Kk * FOUT * FIN; i += 512) {
    int k   = i / (FOUT * FIN);
    int rem = i - k * (FOUT * FIN);
    int o   = rem >> 5;
    int f   = rem & 31;
    sTheta[i] = (__bf16)Theta[((size_t)k * FIN + f) * FOUT + o];
  }

  v8f outAcc[12];                 // [tt(3)][ot(4)] for this wave's m-half
#pragma unroll
  for (int j = 0; j < 12; ++j)
#pragma unroll
    for (int r = 0; r < 8; ++r) outAcc[j][r] = 0.0f;

  __syncthreads();

  const float*  attB  = att + (size_t)b * Nn * Nn;
  const __bf16* xbase = Xt + (size_t)b * TF * Nn;
  const int mm  = tid & 31;       // m row this thread builds in sAt
  const int nnb = tid >> 5;       // n base (0..15), +16 for second element

  for (int k = 0; k < Kk; ++k) {
    const float* chebK = cheb + (size_t)k * Nn * Nn;

    v8f rhs[6];                   // [mt(2)][j(3)]
#pragma unroll
    for (int j = 0; j < 6; ++j)
#pragma unroll
      for (int r = 0; r < 8; ++r) rhs[j][r] = 0.0f;

    // software-pipelined prefetch of the cheb*att tile operands
    float pC0 = chebK[(size_t)nnb * Nn + (m0 + mm)];
    float pA0 = attB [(size_t)nnb * Nn + (m0 + mm)];
    float pC1 = chebK[(size_t)(nnb + 16) * Nn + (m0 + mm)];
    float pA1 = attB [(size_t)(nnb + 16) * Nn + (m0 + mm)];

    // ---- Stage 1: rhs[m, tf] += sum_n (cheb*att)[m,n] * Xt[n, tf] ----
    for (int n0 = 0; n0 < Nn; n0 += 32) {
      sAt[mm * AST + nnb]      = (__bf16)(pC0 * pA0);
      sAt[mm * AST + nnb + 16] = (__bf16)(pC1 * pA1);
      if (n0 + 32 < Nn) {       // issue next step's loads; latency hides under WMMAs
        size_t i0 = (size_t)(n0 + 32 + nnb) * Nn + (m0 + mm);
        size_t i1 = (size_t)(n0 + 48 + nnb) * Nn + (m0 + mm);
        pC0 = chebK[i0]; pA0 = attB[i0];
        pC1 = chebK[i1]; pA1 = attB[i1];
      }
      __syncthreads();

      // A fragments for both m-tiles (contiguous layout: K=h*16+i)
      v16bf aF[2];
#pragma unroll
      for (int mt = 0; mt < 2; ++mt) {
        const v8bf lo = *(const v8bf*)&sAt[(mt * 16 + l16) * AST + h * 16];
        const v8bf hi = *(const v8bf*)&sAt[(mt * 16 + l16) * AST + h * 16 + 8];
#pragma unroll
        for (int i = 0; i < 8; ++i) { aF[mt][i] = lo[i]; aF[mt][i + 8] = hi[i]; }
      }

#pragma unroll
      for (int j = 0; j < 3; ++j) {
        // B fragment (K = h*16 + i): 16 contiguous bf16 along n, 32B-aligned
        int tf = tfbase + j * 16 + l16;
        const v16bf bF = *(const v16bf*)(xbase + (size_t)tf * Nn + n0 + h * 16);
#pragma unroll
        for (int mt = 0; mt < 2; ++mt) {  // one B feeds two WMMAs
          rhs[mt * 3 + j] = __builtin_amdgcn_wmma_f32_16x16x32_bf16(
              false, aF[mt], false, bF, (short)0, rhs[mt * 3 + j], false, false);
        }
      }
      __syncthreads();   // protect sAt before next iteration overwrites it
    }

    // ---- rhs (C-layout regs) -> LDS bf16 [m][tf] ----
#pragma unroll
    for (int mt = 0; mt < 2; ++mt)
#pragma unroll
      for (int j = 0; j < 3; ++j) {
        int tf = tfbase + j * 16 + l16;
#pragma unroll
        for (int r = 0; r < 8; ++r) {
          int m = mt * 16 + r + 8 * h;
          sRhs[m * RST + tf] = (__bf16)rhs[mt * 3 + j][r];
        }
      }
    __syncthreads();

    // ---- Stage 2: out[m,o,t] += rhs[m, t*32+f] @ Theta_k[f,o]  (K=f=32) ----
    {
      const int mt2   = wave >> 3;        // which m-half this wave contracts
      const int tbase = (wave & 7) * 3;   // 3 t values per wave
#pragma unroll
      for (int tt = 0; tt < 3; ++tt) {
        int t = tbase + tt;
        v16bf aF;
        {
          const int row = mt2 * 16 + l16;
          const v8bf lo = *(const v8bf*)&sRhs[row * RST + t * 32 + h * 16];
          const v8bf hi = *(const v8bf*)&sRhs[row * RST + t * 32 + h * 16 + 8];
#pragma unroll
          for (int i = 0; i < 8; ++i) { aF[i] = lo[i]; aF[i + 8] = hi[i]; }
        }
#pragma unroll
        for (int ot = 0; ot < 4; ++ot) {
          int o = ot * 16 + l16;
          // Theta B fragment: 16 contiguous bf16 along f, 32B-aligned
          const v16bf bF = *(const v16bf*)&sTheta[(k * FOUT + o) * FIN + h * 16];
          outAcc[tt * 4 + ot] = __builtin_amdgcn_wmma_f32_16x16x32_bf16(
              false, aF, false, bF, (short)0, outAcc[tt * 4 + ot], false, false);
        }
      }
    }
    __syncthreads();   // before next k reuses sRhs
  }

  // ---- ReLU + coalesced store via LDS, one 16-wide o-slice at a time ----
  const int mt2   = wave >> 3;
  const int tbase = (wave & 7) * 3;
  for (int p = 0; p < 4; ++p) {
#pragma unroll
    for (int tt = 0; tt < 3; ++tt) {
      int t = tbase + tt;
      v8f a = outAcc[tt * 4 + p];
#pragma unroll
      for (int r = 0; r < 8; ++r) {
        int m = mt2 * 16 + r + 8 * h;
        float v = a[r];
        sOutC[(m * 16 + l16) * Tt + t] = v > 0.0f ? v : 0.0f;
      }
    }
    __syncthreads();
    const size_t base = (((size_t)b * Nn + m0) * FOUT + p * 16) * Tt;
    for (int i = tid; i < Mw * 16 * Tt; i += 512) {
      int m   = i / (16 * Tt);
      int rem = i - m * (16 * Tt);           // ol*24 + t
      out[base + (size_t)m * FOUT * Tt + rem] = sOutC[i];
    }
    __syncthreads();
  }
}

// ---------------------------------------------------------------------------
extern "C" void kernel_launch(void* const* d_in, const int* in_sizes, int n_in,
                              void* d_out, int out_size, void* d_ws, size_t ws_size,
                              hipStream_t stream) {
  (void)in_sizes; (void)n_in; (void)out_size; (void)ws_size;
  const float* x     = (const float*)d_in[0];  // (B,N,F_IN,T)
  const float* att   = (const float*)d_in[1];  // (B,N,N)
  const float* cheb  = (const float*)d_in[2];  // (K,N,N)
  const float* Theta = (const float*)d_in[3];  // (K,F_IN,F_OUT)
  __bf16* Xt = (__bf16*)d_ws;                  // (B,TF,N) bf16 = 50.3 MB scratch

  dim3 g1(Nn / 32, TF / 32, Bn);               // 64 x 24 x 16
  xpose_bf16_kernel<<<g1, dim3(32, 8), 0, stream>>>(x, Xt);

  dim3 g2(Nn / Mw, Bn);                        // 64 x 16 workgroups, 512 thr each
  cheb_conv_wmma_kernel<<<g2, 512, 0, stream>>>(att, cheb, Theta, Xt,
                                                (float*)d_out);
}
